// CascadedDifferenceWindowAttention_84267258347841
// MI455X (gfx1250) — matrix operand order repaired and checked by reading, not compiled, not run on hardware
//
#include <hip/hip_runtime.h>
#include <hip/hip_bf16.h>

typedef __bf16 bf16_t;
typedef __bf16 v16bf __attribute__((ext_vector_type(16)));
typedef __bf16 v8bf  __attribute__((ext_vector_type(8)));
typedef __bf16 v4bf  __attribute__((ext_vector_type(4)));
typedef float  v8f   __attribute__((ext_vector_type(8)));

#define TOK ((size_t)(8 * 168 * 168))   // 225792 tokens, divisible by 128 (1764 tiles)

__device__ __forceinline__ v16bf pack16(v8bf lo, v8bf hi) {
  v16bf r;
#pragma unroll
  for (int j = 0; j < 8; ++j) { r[j] = lo[j]; r[j + 8] = hi[j]; }
  return r;
}

// ---------- fp32 [K][N] -> bf16 [N][K] (B-matrix fragment wants contiguous K per column) ----------
__global__ void k_transpose(const float* __restrict__ w, bf16_t* __restrict__ wT, int K, int N) {
  int i = blockIdx.x * blockDim.x + threadIdx.x;
  if (i >= K * N) return;
  int k = i / N, n = i % N;
  wT[(size_t)n * K + k] = (bf16_t)w[i];
}

// ---------- relative position bias: bias[h][p][q] = table[(seq(p)+seq(N-1-q))][h] ----------
__global__ void k_bias(const float* __restrict__ table, float* __restrict__ bias, int WSZ) {
  int N = WSZ * WSZ;
  int i = blockIdx.x * blockDim.x + threadIdx.x;
  if (i >= 8 * N * N) return;
  int q = i % N; int r = i / N; int p = r % N; int h = r / N;
  int sp = (p / WSZ) * (2 * WSZ - 1) + (p % WSZ);
  int qq = N - 1 - q;
  int sq = (qq / WSZ) * (2 * WSZ - 1) + (qq % WSZ);
  bias[i] = table[(size_t)(sp + sq) * 8 + h];
}

// ---------- per-branch input prep: cascade add + windowed difference, fp32 -> bf16 ----------
__global__ void k_prep(const float* __restrict__ x, const float* __restrict__ prev,
                       bf16_t* __restrict__ xb, bf16_t* __restrict__ db, int branch) {
  size_t i = (size_t)blockIdx.x * blockDim.x + threadIdx.x;
  if (i >= TOK * 128) return;
  size_t t = i >> 7; int c = (int)(i & 127);
  int bimg = (int)(t / (168 * 168)); int rem = (int)(t % (168 * 168));
  int yy = rem / 168, xx = rem % 168;
  float v = x[t * 384 + branch * 128 + c];
  if (prev) v += prev[i];
  xb[i] = (bf16_t)v;
  int cy = (yy / 3) * 3 + 1, cx = (xx / 3) * 3 + 1;
  size_t tc = ((size_t)bimg * 168 + cy) * 168 + cx;
  float cv = x[tc * 384 + branch * 128 + c];
  if (prev) cv += prev[tc * 128 + c];
  int r = yy % 3, cc = xx % 3;
  bool sub;
  if (branch == 0)      sub = true;                                   // center_diff: all
  else if (branch == 1) sub = ((r != 1) && (cc != 1)) || (r == 1 && cc == 1); // diag: corners+center
  else                  sub = (r == 1) || (cc == 1);                  // cross: mid row/col
  db[i] = (bf16_t)(sub ? (v - cv) : v);
}

// ---------- fused QKV GEMM: [128 x 128] x 3, A-tile in LDS, B from L2, WMMA bf16 ----------
__global__ __launch_bounds__(256) void k_gemm_qkv(
    const bf16_t* __restrict__ X,
    const bf16_t* __restrict__ wq, const bf16_t* __restrict__ wk, const bf16_t* __restrict__ wv,
    const float* __restrict__ bq, const float* __restrict__ bk, const float* __restrict__ bv,
    bf16_t* __restrict__ qo, bf16_t* __restrict__ ko, bf16_t* __restrict__ vo, float qscale) {
  __shared__ bf16_t At[128 * 128];
  const int tid = threadIdx.x;
  const size_t tile_m = (size_t)blockIdx.x * 128;
  // pull the (L2-resident) weight streams toward this WGP while we stage A
  __builtin_prefetch(wq + threadIdx.x * 64, 0, 1);
  __builtin_prefetch(wk + threadIdx.x * 64, 0, 1);
  __builtin_prefetch(wv + threadIdx.x * 64, 0, 1);
  {
    const v8bf* src = (const v8bf*)(X + tile_m * 128);
    v8bf* dst = (v8bf*)At;
#pragma unroll
    for (int it = 0; it < 8; ++it) dst[tid + it * 256] = src[tid + it * 256];
  }
  __syncthreads();
  const int wave = tid >> 5, lane = tid & 31;
  const int ml = lane & 15, kh = lane >> 4;
  const v8f vzero = {};

  // A fragments depend only on ks -- load once, reuse for q/k/v (saves 2/3 of LDS reads)
  v16bf afrag[4];
#pragma unroll
  for (int ks = 0; ks < 4; ++ks) {
    const bf16_t* ar = At + (wave * 16 + ml) * 128 + ks * 32;
    afrag[ks] = pack16(*(const v8bf*)(ar + kh * 16), *(const v8bf*)(ar + kh * 16 + 8));
  }

#pragma unroll
  for (int mat = 0; mat < 3; ++mat) {
    const bf16_t* wsel = (mat == 0) ? wq : ((mat == 1) ? wk : wv);
    const float*  bsel = (mat == 0) ? bq : ((mat == 1) ? bk : bv);
    bf16_t*       osel = (mat == 0) ? qo : ((mat == 1) ? ko : vo);
    const float sc = (mat == 0) ? qscale : 1.0f;
    v8f acc[8];
#pragma unroll
    for (int nt = 0; nt < 8; ++nt) acc[nt] = vzero;
#pragma unroll
    for (int ks = 0; ks < 4; ++ks) {
#pragma unroll
      for (int nt = 0; nt < 8; ++nt) {
        const bf16_t* br = wsel + (size_t)(nt * 16 + ml) * 128 + ks * 32 + kh * 16;
        v16bf bfrag = pack16(*(const v8bf*)br, *(const v8bf*)(br + 8));
        acc[nt] = __builtin_amdgcn_wmma_f32_16x16x32_bf16(false, afrag[ks], false, bfrag,
                                                          (short)0, acc[nt], false, false);
      }
    }
    const size_t rbase = tile_m + (size_t)wave * 16 + kh * 8;
#pragma unroll
    for (int nt = 0; nt < 8; ++nt) {
      const int n = nt * 16 + ml;
      const float bvv = bsel[n];
#pragma unroll
      for (int i = 0; i < 8; ++i)
        osel[(rbase + i) * 128 + n] = (bf16_t)((acc[nt][i] + bvv) * sc);
    }
  }
}

// ---------- output projection GEMM, fp32 accumulate into branch output ----------
__global__ __launch_bounds__(256) void k_gemm_proj(
    const bf16_t* __restrict__ X, const bf16_t* __restrict__ wp,
    const float* __restrict__ bp, float* __restrict__ out, int accumulate) {
  __shared__ bf16_t At[128 * 128];
  const int tid = threadIdx.x;
  const size_t tile_m = (size_t)blockIdx.x * 128;
  __builtin_prefetch(wp + threadIdx.x * 64, 0, 1);
  {
    const v8bf* src = (const v8bf*)(X + tile_m * 128);
    v8bf* dst = (v8bf*)At;
#pragma unroll
    for (int it = 0; it < 8; ++it) dst[tid + it * 256] = src[tid + it * 256];
  }
  __syncthreads();
  const int wave = tid >> 5, lane = tid & 31;
  const int ml = lane & 15, kh = lane >> 4;
  const v8f vzero = {};
  v8f acc[8];
#pragma unroll
  for (int nt = 0; nt < 8; ++nt) acc[nt] = vzero;
#pragma unroll
  for (int ks = 0; ks < 4; ++ks) {
    const bf16_t* ar = At + (wave * 16 + ml) * 128 + ks * 32;
    v16bf afrag = pack16(*(const v8bf*)(ar + kh * 16), *(const v8bf*)(ar + kh * 16 + 8));
#pragma unroll
    for (int nt = 0; nt < 8; ++nt) {
      const bf16_t* br = wp + (size_t)(nt * 16 + ml) * 128 + ks * 32 + kh * 16;
      v16bf bfrag = pack16(*(const v8bf*)br, *(const v8bf*)(br + 8));
      acc[nt] = __builtin_amdgcn_wmma_f32_16x16x32_bf16(false, afrag, false, bfrag,
                                                        (short)0, acc[nt], false, false);
    }
  }
  const size_t rbase = tile_m + (size_t)wave * 16 + kh * 8;
#pragma unroll
  for (int nt = 0; nt < 8; ++nt) {
    const int n = nt * 16 + ml;
    const float bvv = bp[n];
#pragma unroll
    for (int i = 0; i < 8; ++i) {
      const size_t idx = (rbase + i) * 128 + n;
      float vv = acc[nt][i] + bvv;
      if (accumulate) vv += out[idx];
      out[idx] = vv;
    }
  }
}

// ---------- window attention core: flash-style online softmax, one thread per (win,q,head) ----------
template <int WSZ>
__global__ void k_attn(const bf16_t* __restrict__ q, const bf16_t* __restrict__ k,
                       const bf16_t* __restrict__ v, const float* __restrict__ bias,
                       bf16_t* __restrict__ ao, int nwin) {
  const int N = WSZ * WSZ;
  size_t item = (size_t)blockIdx.x * blockDim.x + threadIdx.x;
  size_t total = (size_t)8 * nwin * nwin * N * 8;
  if (item >= total) return;
  int h = (int)(item & 7);
  size_t r1 = item >> 3;
  int qp = (int)(r1 % N);
  size_t win = r1 / N;
  int bimg = (int)(win / (nwin * nwin));
  int wrem = (int)(win % (nwin * nwin));
  int wy = wrem / nwin, wx = wrem % nwin;
  const size_t rowbase = ((size_t)bimg * 168 + wy * WSZ) * 168 + wx * WSZ;
  const size_t tq = rowbase + (size_t)(qp / WSZ) * 168 + (qp % WSZ);
  float qr[16];
  {
    const v8bf q0 = *(const v8bf*)(q + tq * 128 + h * 16);
    const v8bf q1 = *(const v8bf*)(q + tq * 128 + h * 16 + 8);
#pragma unroll
    for (int j = 0; j < 8; ++j) { qr[j] = (float)q0[j]; qr[j + 8] = (float)q1[j]; }
  }
  const float* bi = bias + ((size_t)h * N + qp) * N;
  float mrun = -3.0e38f, srun = 0.f, acc[16];
#pragma unroll
  for (int j = 0; j < 16; ++j) acc[j] = 0.f;
  for (int kp = 0; kp < N; ++kp) {
    const size_t tk = rowbase + (size_t)(kp / WSZ) * 168 + (kp % WSZ);
    const v8bf k0 = *(const v8bf*)(k + tk * 128 + h * 16);
    const v8bf k1 = *(const v8bf*)(k + tk * 128 + h * 16 + 8);
    float d = 0.f;
#pragma unroll
    for (int j = 0; j < 8; ++j) d += qr[j] * (float)k0[j] + qr[j + 8] * (float)k1[j];
    d += bi[kp];
    const float mn = fmaxf(mrun, d);
    const float f = __expf(mrun - mn);
    const float p = __expf(d - mn);
    const v8bf v0 = *(const v8bf*)(v + tk * 128 + h * 16);
    const v8bf v1 = *(const v8bf*)(v + tk * 128 + h * 16 + 8);
#pragma unroll
    for (int j = 0; j < 8; ++j) {
      acc[j] = acc[j] * f + p * (float)v0[j];
      acc[j + 8] = acc[j + 8] * f + p * (float)v1[j];
    }
    srun = srun * f + p;
    mrun = mn;
  }
  const float inv = 1.f / srun;
  bf16_t* op = ao + tq * 128 + h * 16;
#pragma unroll
  for (int j = 0; j < 16; ++j) op[j] = (bf16_t)(acc[j] * inv);
}

// ---------- merge GEMM: concat(out1,out2,out3)[T x 384] @ mergeW[384x384] + b, WMMA ----------
__global__ __launch_bounds__(256) void k_gemm_merge(
    const float* __restrict__ o0, const float* __restrict__ o1, const float* __restrict__ o2,
    const bf16_t* __restrict__ mT, const float* __restrict__ mb, float* __restrict__ Y) {
  __shared__ bf16_t At[128 * 128];
  const int tid = threadIdx.x;
  const size_t tile_m = (size_t)blockIdx.x * 128;
  const int nchunk = blockIdx.y;
  __builtin_prefetch(mT + (size_t)nchunk * 128 * 384 + threadIdx.x * 576, 0, 1);
  const int wave = tid >> 5, lane = tid & 31, ml = lane & 15, kh = lane >> 4;
  const v8f vzero = {};
  v8f acc[8];
#pragma unroll
  for (int nt = 0; nt < 8; ++nt) acc[nt] = vzero;
  for (int kc = 0; kc < 3; ++kc) {
    const float* src = (kc == 0) ? o0 : ((kc == 1) ? o1 : o2);
    __syncthreads();
    for (int i = tid * 4; i < 128 * 128; i += 256 * 4) {
      float4 f = *(const float4*)(src + tile_m * 128 + i);
      v4bf t; t[0] = (bf16_t)f.x; t[1] = (bf16_t)f.y; t[2] = (bf16_t)f.z; t[3] = (bf16_t)f.w;
      *(v4bf*)(At + i) = t;
    }
    __syncthreads();
#pragma unroll
    for (int ks = 0; ks < 4; ++ks) {
      const bf16_t* ar = At + (wave * 16 + ml) * 128 + ks * 32;
      v16bf afrag = pack16(*(const v8bf*)(ar + kh * 16), *(const v8bf*)(ar + kh * 16 + 8));
#pragma unroll
      for (int nt = 0; nt < 8; ++nt) {
        const int n = nchunk * 128 + nt * 16 + ml;
        const bf16_t* br = mT + (size_t)n * 384 + kc * 128 + ks * 32 + kh * 16;
        v16bf bfrag = pack16(*(const v8bf*)br, *(const v8bf*)(br + 8));
        acc[nt] = __builtin_amdgcn_wmma_f32_16x16x32_bf16(false, afrag, false, bfrag,
                                                          (short)0, acc[nt], false, false);
      }
    }
  }
  const size_t rbase = tile_m + (size_t)wave * 16 + kh * 8;
#pragma unroll
  for (int nt = 0; nt < 8; ++nt) {
    const int n = nchunk * 128 + nt * 16 + ml;
    const float bvv = mb[n];
#pragma unroll
    for (int i = 0; i < 8; ++i) Y[(rbase + i) * 384 + n] = acc[nt][i] + bvv;
  }
}

extern "C" void kernel_launch(void* const* d_in, const int* in_sizes, int n_in,
                              void* d_out, int out_size, void* d_ws, size_t ws_size,
                              hipStream_t stream) {
  (void)in_sizes; (void)n_in; (void)out_size; (void)ws_size;
  const float* x = (const float*)d_in[0];
  // params order per setup_inputs: center,diag,cross,reg1,reg2,reg3 each {qw,qb,kw,kb,vw,vb,pw,pb,table}
  auto P = [&](int pi, int j) -> const float* { return (const float*)d_in[1 + pi * 9 + j]; };
  const float* mergew = (const float*)d_in[55];
  const float* mergeb = (const float*)d_in[56];

  char* ws = (char*)d_ws;
  size_t off = 0;
  auto alloc = [&](size_t bytes) -> char* {
    char* r = ws + off; off = (off + bytes + 255) & ~(size_t)255; return r;
  };
  bf16_t* wT    = (bf16_t*)alloc(24 * 16384 * 2);          // 6 param sets x {q,k,v,p} transposed bf16
  bf16_t* mT    = (bf16_t*)alloc(384 * 384 * 2);
  float*  bias3 = (float*)alloc(3 * 8 * 81 * 4);
  float*  bias7 = (float*)alloc(3 * 8 * 2401 * 4);
  bf16_t* xb    = (bf16_t*)alloc(TOK * 128 * 2);
  bf16_t* db    = (bf16_t*)alloc(TOK * 128 * 2);
  bf16_t* qbuf  = (bf16_t*)alloc(TOK * 128 * 2);
  bf16_t* kbuf  = (bf16_t*)alloc(TOK * 128 * 2);
  bf16_t* vbuf  = (bf16_t*)alloc(TOK * 128 * 2);
  bf16_t* ao    = (bf16_t*)alloc(TOK * 128 * 2);
  float*  outp[3];
  for (int b = 0; b < 3; ++b) outp[b] = (float*)alloc(TOK * 128 * 4);

  auto wTp = [&](int pi, int mat) -> bf16_t* { return wT + (size_t)(pi * 4 + mat) * 16384; };

  // --- weight prep ---
  for (int pi = 0; pi < 6; ++pi)
    for (int mat = 0; mat < 4; ++mat)
      k_transpose<<<64, 256, 0, stream>>>(P(pi, mat * 2), wTp(pi, mat), 128, 128);
  k_transpose<<<576, 256, 0, stream>>>(mergew, mT, 384, 384);
  for (int pi = 0; pi < 3; ++pi)
    k_bias<<<3, 256, 0, stream>>>(P(pi, 8), bias3 + (size_t)pi * 8 * 81, 3);
  for (int pi = 3; pi < 6; ++pi)
    k_bias<<<76, 256, 0, stream>>>(P(pi, 8), bias7 + (size_t)(pi - 3) * 8 * 2401, 7);

  const int mtiles = (int)(TOK / 128);            // 1764
  const int prep_blocks = (int)(TOK * 128 / 256); // 112896
  const int attn_blocks = 7056;                   // 1,806,336 work items / 256 (both window sizes)
  const float qscale = 0.25f;                     // hd^-0.5, hd=16

  for (int br = 0; br < 3; ++br) {
    const int pd = br, pr = br + 3;
    const float* prev = (br == 0) ? nullptr : outp[br - 1];
    k_prep<<<prep_blocks, 256, 0, stream>>>(x, prev, xb, db, br);

    // 3x3 difference-window attention path
    k_gemm_qkv<<<mtiles, 256, 0, stream>>>(db, wTp(pd, 0), wTp(pd, 1), wTp(pd, 2),
                                           P(pd, 1), P(pd, 3), P(pd, 5),
                                           qbuf, kbuf, vbuf, qscale);
    k_attn<3><<<attn_blocks, 256, 0, stream>>>(qbuf, kbuf, vbuf,
                                               bias3 + (size_t)br * 8 * 81, ao, 56);
    k_gemm_proj<<<mtiles, 256, 0, stream>>>(ao, wTp(pd, 3), P(pd, 7), outp[br], 0);

    // 7x7 regular-window attention path (accumulates into out_b)
    k_gemm_qkv<<<mtiles, 256, 0, stream>>>(xb, wTp(pr, 0), wTp(pr, 1), wTp(pr, 2),
                                           P(pr, 1), P(pr, 3), P(pr, 5),
                                           qbuf, kbuf, vbuf, qscale);
    k_attn<7><<<attn_blocks, 256, 0, stream>>>(qbuf, kbuf, vbuf,
                                               bias7 + (size_t)br * 8 * 2401, ao, 24);
    k_gemm_proj<<<mtiles, 256, 0, stream>>>(ao, wTp(pr, 3), P(pr, 7), outp[br], 1);
  }

  // final merge: concat(out1,out2,out3) @ merge_w + merge_b
  k_gemm_merge<<<dim3(mtiles, 3), 256, 0, stream>>>(outp[0], outp[1], outp[2],
                                                    mT, mergeb, (float*)d_out);
}
